// PointsFusion_60945585931017
// MI455X (gfx1250) — matrix operand order repaired and checked by reading, not compiled, not run on hardware
//
#include <hip/hip_runtime.h>
#include <hip/hip_bf16.h>
#include <math.h>

typedef __attribute__((ext_vector_type(16))) _Float16 v16h;
typedef __attribute__((ext_vector_type(8)))  float    v8f;
typedef __attribute__((ext_vector_type(4)))  unsigned int u32x4;
typedef __attribute__((ext_vector_type(8)))  int i32x8;
typedef __attribute__((ext_vector_type(4)))  int i32x4;

namespace {
constexpr int B_  = 2;
constexpr int N_  = 4096;   // queries per batch
constexpr int P_  = 8192;   // candidate points per batch (2N)
constexpr int C_  = 128;    // feature channels
constexpr int K_  = 32;     // neighbors
constexpr int BS_ = B_ * N_ * K_;   // 262144 spatial samples for the MLP
constexpr float EPS_ = 1e-5f;
}

// ---------------------------------------------------------------------------
// 1) Build concatenated point / feature tables in gather-friendly layouts
//    pts2[b][p][3], f2[b][p][128]  (row-major, point-major)
// ---------------------------------------------------------------------------
__global__ void build_f2_kernel(const float* __restrict__ fe1,
                                const float* __restrict__ fe2,
                                float* __restrict__ f2) {
  size_t i = (size_t)blockIdx.x * blockDim.x + threadIdx.x; // B*P*128
  int c = (int)(i % C_);
  size_t bp = i / C_;
  int p = (int)(bp % P_);
  int b = (int)(bp / P_);
  float v = (p < N_) ? fe1[((size_t)b * C_ + c) * N_ + p]
                     : fe2[((size_t)b * C_ + c) * N_ + (p - N_)];
  f2[i] = v;
}

__global__ void build_pts2_kernel(const float* __restrict__ po1,
                                  const float* __restrict__ po2,
                                  float* __restrict__ pts2) {
  int i = blockIdx.x * blockDim.x + threadIdx.x; // B*P*3
  if (i >= B_ * P_ * 3) return;
  int d = i % 3;
  int bp = i / 3;
  int p = bp % P_;
  int b = bp / P_;
  pts2[i] = (p < N_) ? po1[(b * 3 + d) * N_ + p]
                     : po2[(b * 3 + d) * N_ + (p - N_)];
}

// p1[b][m][3], f1[b][m][128] gathered through randidx
__global__ void gather_kernel(const float* __restrict__ pts2,
                              const float* __restrict__ f2,
                              const int* __restrict__ ridx,
                              float* __restrict__ p1,
                              float* __restrict__ f1) {
  size_t i = (size_t)blockIdx.x * blockDim.x + threadIdx.x; // B*N*128
  int c = (int)(i % C_);
  size_t bm = i / C_;
  int m = (int)(bm % N_);
  int b = (int)(bm / N_);
  int r = ridx[m];
  f1[i] = f2[((size_t)b * P_ + r) * C_ + c];
  if (c < 3) p1[bm * 3 + c] = pts2[((size_t)b * P_ + r) * 3 + c];
}

// ---------------------------------------------------------------------------
// 2) Exact KNN (top-32 smallest squared distance) — per-thread insertion list,
//    candidate points streamed through LDS in 24KB chunks.
// ---------------------------------------------------------------------------
#define KNN_CHUNK 2048
__global__ void knn_kernel(const float* __restrict__ pts2,
                           const float* __restrict__ p1,
                           int* __restrict__ nnidx) {
  __shared__ float spts[KNN_CHUNK * 3];
  int gq = blockIdx.x * blockDim.x + threadIdx.x;   // query id in [0, B*N)
  int b = gq / N_;
  float qx = p1[gq * 3 + 0];
  float qy = p1[gq * 3 + 1];
  float qz = p1[gq * 3 + 2];
  float bd[K_];
  int   bi[K_];
#pragma unroll
  for (int i = 0; i < K_; ++i) { bd[i] = 3.4e38f; bi[i] = 0; }

  for (int c0 = 0; c0 < P_; c0 += KNN_CHUNK) {
    __syncthreads();
    for (int t = threadIdx.x; t < KNN_CHUNK * 3; t += blockDim.x)
      spts[t] = pts2[((size_t)b * P_ + c0) * 3 + t];
    __syncthreads();
    for (int j = 0; j < KNN_CHUNK; ++j) {
      float dx = qx - spts[j * 3 + 0];
      float dy = qy - spts[j * 3 + 1];
      float dz = qz - spts[j * 3 + 2];
      float d = dx * dx + dy * dy + dz * dz;
      if (d < bd[K_ - 1]) {
        int pos = K_ - 1;
        while (pos > 0 && bd[pos - 1] > d) {
          bd[pos] = bd[pos - 1];
          bi[pos] = bi[pos - 1];
          --pos;
        }
        bd[pos] = d;
        bi[pos] = c0 + j;
      }
    }
  }
  for (int i = 0; i < K_; ++i) nnidx[(size_t)gq * K_ + i] = bi[i];
}

// ---------------------------------------------------------------------------
// 3) Build MLP input X0[s][32] (f16, channels: rx,ry,rz,dist,fdot, 27 zeros)
// ---------------------------------------------------------------------------
__global__ void newf_kernel(const float* __restrict__ pts2,
                            const float* __restrict__ f2,
                            const float* __restrict__ p1,
                            const float* __restrict__ f1,
                            const int* __restrict__ nnidx,
                            _Float16* __restrict__ X0) {
  int s = blockIdx.x * blockDim.x + threadIdx.x;  // [0, BS)
  int k = s % K_;
  int q = s / K_;            // (b*N + m)
  int b = q / N_;
  int id = nnidx[(size_t)q * K_ + k];
  const float* np = pts2 + ((size_t)b * P_ + id) * 3;
  float rx = np[0] - p1[q * 3 + 0];
  float ry = np[1] - p1[q * 3 + 1];
  float rz = np[2] - p1[q * 3 + 2];
  float dist = sqrtf(rx * rx + ry * ry + rz * rz);
  const float4* fa = (const float4*)(f1 + (size_t)q * C_);
  const float4* fb = (const float4*)(f2 + ((size_t)b * P_ + id) * C_);
  float acc = 0.f;
#pragma unroll 8
  for (int c = 0; c < C_ / 4; ++c) {
    float4 a = fa[c], x = fb[c];
    acc += a.x * x.x + a.y * x.y + a.z * x.z + a.w * x.w;
  }
  union { _Float16 h[32]; uint4 q4[4]; } o;
#pragma unroll
  for (int j = 0; j < 32; ++j) o.h[j] = (_Float16)0.f;
  o.h[0] = (_Float16)rx;  o.h[1] = (_Float16)ry;  o.h[2] = (_Float16)rz;
  o.h[3] = (_Float16)dist; o.h[4] = (_Float16)acc;
  uint4* dst = (uint4*)(X0 + (size_t)s * 32);
#pragma unroll
  for (int j = 0; j < 4; ++j) dst[j] = o.q4[j];
}

// ---------------------------------------------------------------------------
// 4) Weight convert: f32 [COUT][CINr] -> f16 [COUT][CINp] (zero padded)
// ---------------------------------------------------------------------------
__global__ void wconv_kernel(const float* __restrict__ w,
                             _Float16* __restrict__ wh,
                             int COUT, int CINr, int CINp) {
  int i = blockIdx.x * blockDim.x + threadIdx.x;
  if (i >= COUT * CINp) return;
  int o = i / CINp, c = i % CINp;
  wh[i] = (c < CINr) ? (_Float16)w[o * CINr + c] : (_Float16)0.f;
}

// ---------------------------------------------------------------------------
// 5) WMMA GEMM: Y[s][o] = sum_c W[o][c] * X[s][c]   (f16 in, f32 acc, f16 out)
//    Workgroup: 8 waves over output channels; each wave computes a 16(o)x64(s)
//    tile (4 accumulators, A-fragment reused across 4 WMMAs per K-step).
//    The 64xCIN f16 X-tile is CONTIGUOUS in memory (s-major layout) and is
//    staged into LDS with one Tensor-Data-Mover transfer issued by wave 0.
// ---------------------------------------------------------------------------
template <int CIN, int COUT>
__global__ void gemm_kernel(const _Float16* __restrict__ X,
                            const _Float16* __restrict__ W,
                            _Float16* __restrict__ Y) {
  __shared__ _Float16 bt[64 * CIN];
  const int s0 = blockIdx.x * 64;
  const int obase = blockIdx.y * 128;
  const int t = threadIdx.x;

  if (t < 32) {
    // --- TDM descriptor (D#): 1-D contiguous tile of 64*CIN halves ---
    unsigned long long gaddr =
        (unsigned long long)(uintptr_t)(X + (size_t)s0 * CIN);
    unsigned int laddr = (unsigned int)(uintptr_t)bt;  // LDS byte offset
    const unsigned int td0   = (unsigned int)((size_t)BS_ * CIN / 4); // tensor len (8B units)
    const unsigned int tile0 = (unsigned int)(64 * CIN / 4);          // tile len  (8B units)
    u32x4 g0 = {};
    g0[0] = 1u;                                  // count=1, user mode
    g0[1] = laddr;                               // lds_addr
    g0[2] = (unsigned int)gaddr;                 // global_addr[31:0]
    g0[3] = (unsigned int)(gaddr >> 32) | (2u << 30);  // addr[56:32] | type=2
    i32x8 g1 = {};
    g1[0] = (int)(3u << 16);                     // data_size = 3 (8 bytes)
    g1[1] = (int)((td0 & 0xFFFFu) << 16);        // tensor_dim0[15:0]
    g1[2] = (int)((td0 >> 16) | (1u << 16));     // tensor_dim0[31:16] | tensor_dim1=1
    g1[3] = (int)(tile0 << 16);                  // tile_dim0
    g1[5] = (int)td0;                            // tensor_dim0_stride[31:0]
    i32x4 g2 = {};
    i32x4 g3 = {};
    i32x8 g4 = {};
    __builtin_amdgcn_tensor_load_to_lds(g0, g1, g2, g3, g4, 0);
    __builtin_amdgcn_s_wait_tensorcnt(0);
  }
  __syncthreads();

  const int wave = t >> 5, lane = t & 31;
  const int o0 = obase + wave * 16;
  const int m  = lane & 15;   // A row (M) == D column (N)
  const int kh = lane >> 4;   // lane-half selects K sub-range / D row-half
  v8f acc[4] = {{}, {}, {}, {}};

#pragma unroll
  for (int k0 = 0; k0 < CIN; k0 += 32) {
    union { v16h v; uint4 q[2]; } A;
    // A (16x32): halves 0..7 -> K = k0 + kh*8 + i ; halves 8..15 -> +16
    const _Float16* wrow = W + (size_t)(o0 + m) * CIN + k0 + kh * 8;
    A.q[0] = *(const uint4*)(wrow);
    A.q[1] = *(const uint4*)(wrow + 16);
#pragma unroll
    for (int j = 0; j < 4; ++j) {
      union { v16h v; uint4 q[2]; } Bf;
      // B (32x16): lane column n=m, halves 0..7 -> K = k0 + kh*8 + i ; halves 8..15 -> +16
      const _Float16* brow = &bt[(j * 16 + m) * CIN + k0 + kh * 8];
      Bf.q[0] = *(const uint4*)(brow);
      Bf.q[1] = *(const uint4*)(brow + 16);
      acc[j] = __builtin_amdgcn_wmma_f32_16x16x32_f16(
          false, A.v, false, Bf.v, (short)0, acc[j], false, false);
    }
  }

  // D: lane (n=m, mh=kh) holds channels o0 + kh*8 + e -> one b128 store each
#pragma unroll
  for (int j = 0; j < 4; ++j) {
    union { _Float16 h[8]; uint4 q; } o;
#pragma unroll
    for (int e = 0; e < 8; ++e) o.h[e] = (_Float16)acc[j][e];
    *(uint4*)(Y + (size_t)(s0 + j * 16 + m) * COUT + o0 + kh * 8) = o.q;
  }
}

// ---------------------------------------------------------------------------
// 6) Deterministic BN statistics: per-WG partial sums -> ordered finalize
// ---------------------------------------------------------------------------
template <int COUT>
__global__ void stats_kernel(const _Float16* __restrict__ Y,
                             float* __restrict__ partials) {
  const int c = threadIdx.x;       // blockDim.x == COUT
  const int wg = blockIdx.x;       // 256 workgroups
  const int SCH = BS_ / 256;
  size_t s0 = (size_t)wg * SCH;
  float s = 0.f, ss = 0.f;
  for (int i = 0; i < SCH; ++i) {
    float v = (float)Y[(s0 + i) * COUT + c];
    s += v;
    ss += v * v;
  }
  partials[((size_t)wg * COUT + c) * 2 + 0] = s;
  partials[((size_t)wg * COUT + c) * 2 + 1] = ss;
}

template <int COUT>
__global__ void finalize_kernel(const float* __restrict__ partials,
                                const float* __restrict__ g,
                                const float* __restrict__ be,
                                float* __restrict__ np) {
  const int c = threadIdx.x;
  float s = 0.f, ss = 0.f;
  for (int w = 0; w < 256; ++w) {
    s  += partials[((size_t)w * COUT + c) * 2 + 0];
    ss += partials[((size_t)w * COUT + c) * 2 + 1];
  }
  const float inv = 1.0f / (float)BS_;
  float mean = s * inv;
  float var = ss * inv - mean * mean;
  float sc = g[c] * rsqrtf(var + EPS_);
  np[2 * c + 0] = sc;
  np[2 * c + 1] = be[c] - mean * sc;
}

// ---------------------------------------------------------------------------
// 7) In-place BN affine + ReLU on f16 activations
// ---------------------------------------------------------------------------
template <int COUT>
__global__ void normrelu_kernel(_Float16* __restrict__ Y,
                                const float* __restrict__ np) {
  size_t i8 = ((size_t)blockIdx.x * blockDim.x + threadIdx.x) * 8;
  int c0 = (int)(i8 % COUT);
  union { _Float16 h[8]; uint4 q; } v;
  v.q = *(uint4*)(Y + i8);
#pragma unroll
  for (int j = 0; j < 8; ++j) {
    float x = (float)v.h[j];
    x = np[2 * (c0 + j) + 0] * x + np[2 * (c0 + j) + 1];
    v.h[j] = (_Float16)fmaxf(x, 0.f);
  }
  *(uint4*)(Y + i8) = v.q;
}

// ---------------------------------------------------------------------------
// 8) Channel-max of BN+ReLU output of the last layer (COUT = 256)
// ---------------------------------------------------------------------------
__global__ void maxred_kernel(const _Float16* __restrict__ Y,
                              const float* __restrict__ np,
                              float* __restrict__ cmax) {
  size_t s = (size_t)blockIdx.x * blockDim.x + threadIdx.x;
  const _Float16* row = Y + s * 256;
  float mx = -3.4e38f;
  for (int c = 0; c < 256; c += 8) {
    union { _Float16 h[8]; uint4 q; } v;
    v.q = *(const uint4*)(row + c);
#pragma unroll
    for (int j = 0; j < 8; ++j) {
      float x = (float)v.h[j];
      x = fmaxf(np[2 * (c + j) + 0] * x + np[2 * (c + j) + 1], 0.f);
      mx = fmaxf(mx, x);
    }
  }
  cmax[s] = mx;
}

// ---------------------------------------------------------------------------
// 9) Softmax over K (wave32 shuffle) + weighted gather fusion
//    out (B,131,N): ch 0..2 = wts_a * neighbor points, 3..130 = wts_b * feats
// ---------------------------------------------------------------------------
__global__ void fusion_kernel(const float* __restrict__ cmax_a,
                              const float* __restrict__ cmax_b,
                              const int* __restrict__ nnidx,
                              const float* __restrict__ pts2,
                              const float* __restrict__ f2,
                              float* __restrict__ out) {
  __shared__ float wA[8][K_];
  __shared__ float wB[8][K_];
  __shared__ int   sid[8][K_];
  const int wave = threadIdx.x >> 5, lane = threadIdx.x & 31;
  const int q = blockIdx.x * 8 + wave;     // [0, B*N)
  const int b = q / N_, m = q % N_;
  const size_t s = (size_t)q * K_ + lane;

  float ca = cmax_a[s], cb = cmax_b[s];
  float amax = ca, bmax = cb;
#pragma unroll
  for (int off = 16; off > 0; off >>= 1) {
    amax = fmaxf(amax, __shfl_xor(amax, off, 32));
    bmax = fmaxf(bmax, __shfl_xor(bmax, off, 32));
  }
  float ea = expf(ca - amax), eb = expf(cb - bmax);
  float sa = ea, sb = eb;
#pragma unroll
  for (int off = 16; off > 0; off >>= 1) {
    sa += __shfl_xor(sa, off, 32);
    sb += __shfl_xor(sb, off, 32);
  }
  wA[wave][lane] = ea / sa;
  wB[wave][lane] = eb / sb;
  sid[wave][lane] = nnidx[s];
  __syncthreads();

  float accf[4] = {0.f, 0.f, 0.f, 0.f};
  float accp = 0.f;
  for (int k = 0; k < K_; ++k) {
    int id = sid[wave][k];
    float wb = wB[wave][k];
    const float* fr = f2 + ((size_t)b * P_ + id) * C_;
#pragma unroll
    for (int j = 0; j < 4; ++j) accf[j] += wb * fr[lane + 32 * j];
    if (lane < 3) accp += wA[wave][k] * pts2[((size_t)b * P_ + id) * 3 + lane];
  }
  float* ob = out + (size_t)b * 131 * N_;
#pragma unroll
  for (int j = 0; j < 4; ++j) ob[(size_t)(3 + lane + 32 * j) * N_ + m] = accf[j];
  if (lane < 3) ob[(size_t)lane * N_ + m] = accp;
}

// ---------------------------------------------------------------------------
// Host-side orchestration
// ---------------------------------------------------------------------------
extern "C" void kernel_launch(void* const* d_in, const int* in_sizes, int n_in,
                              void* d_out, int out_size, void* d_ws, size_t ws_size,
                              hipStream_t stream) {
  (void)in_sizes; (void)n_in; (void)out_size; (void)ws_size;

  const float* points1   = (const float*)d_in[0];
  const float* points2   = (const float*)d_in[1];
  const float* features1 = (const float*)d_in[2];
  const float* features2 = (const float*)d_in[3];
  const int*   randidx   = (const int*)d_in[4];
  // d_in[5] = k (always 32, compile-time)
  const float* Wf[2][3] = {
      {(const float*)d_in[6],  (const float*)d_in[10], (const float*)d_in[14]},
      {(const float*)d_in[18], (const float*)d_in[22], (const float*)d_in[26]}};
  const float* Gf[2][3] = {
      {(const float*)d_in[8],  (const float*)d_in[12], (const float*)d_in[16]},
      {(const float*)d_in[20], (const float*)d_in[24], (const float*)d_in[28]}};
  const float* Bef[2][3] = {
      {(const float*)d_in[9],  (const float*)d_in[13], (const float*)d_in[17]},
      {(const float*)d_in[21], (const float*)d_in[25], (const float*)d_in[29]}};

  float* out = (float*)d_out;

  // ---- workspace carve-up (256B aligned) ----
  char* ws = (char*)d_ws;
  size_t off = 0;
  auto carve = [&](size_t bytes) -> char* {
    char* p = ws + off;
    off += (bytes + 255) & ~(size_t)255;
    return p;
  };
  float*    pts2  = (float*)carve((size_t)B_ * P_ * 3 * 4);
  float*    f2    = (float*)carve((size_t)B_ * P_ * C_ * 4);
  float*    p1    = (float*)carve((size_t)B_ * N_ * 3 * 4);
  float*    f1    = (float*)carve((size_t)B_ * N_ * C_ * 4);
  int*      nnidx = (int*)carve((size_t)B_ * N_ * K_ * 4);
  _Float16* X0    = (_Float16*)carve((size_t)BS_ * 32 * 2);
  _Float16* YA    = (_Float16*)carve((size_t)BS_ * 128 * 2);
  _Float16* YB    = (_Float16*)carve((size_t)BS_ * 128 * 2);
  _Float16* YC    = (_Float16*)carve((size_t)BS_ * 256 * 2);
  _Float16* Wh[2][3];
  for (int br = 0; br < 2; ++br) {
    Wh[br][0] = (_Float16*)carve((size_t)128 * 32 * 2);
    Wh[br][1] = (_Float16*)carve((size_t)128 * 128 * 2);
    Wh[br][2] = (_Float16*)carve((size_t)256 * 128 * 2);
  }
  float* partials = (float*)carve((size_t)256 * 256 * 2 * 4);
  float* normp    = (float*)carve((size_t)2 * 256 * 4);
  float* cmax_a   = (float*)carve((size_t)BS_ * 4);
  float* cmax_b   = (float*)carve((size_t)BS_ * 4);

  // ---- 1) layout transforms ----
  build_f2_kernel<<<(B_ * P_ * C_) / 256, 256, 0, stream>>>(features1, features2, f2);
  build_pts2_kernel<<<(B_ * P_ * 3 + 255) / 256, 256, 0, stream>>>(points1, points2, pts2);
  gather_kernel<<<(B_ * N_ * C_) / 256, 256, 0, stream>>>(pts2, f2, randidx, p1, f1);

  // ---- 2) KNN ----
  knn_kernel<<<(B_ * N_) / 256, 256, 0, stream>>>(pts2, p1, nnidx);

  // ---- 3) MLP input ----
  newf_kernel<<<BS_ / 256, 256, 0, stream>>>(pts2, f2, p1, f1, nnidx, X0);

  // ---- 4) weight conversion (zero-padded f16) ----
  const int cinr[3] = {5, 128, 128};
  const int cinp[3] = {32, 128, 128};
  const int cout[3] = {128, 128, 256};
  for (int br = 0; br < 2; ++br)
    for (int l = 0; l < 3; ++l) {
      int n = cout[l] * cinp[l];
      wconv_kernel<<<(n + 255) / 256, 256, 0, stream>>>(Wf[br][l], Wh[br][l],
                                                        cout[l], cinr[l], cinp[l]);
    }

  // ---- 5) MLP branches: WMMA GEMM + global-BN + ReLU ----
  for (int br = 0; br < 2; ++br) {
    // L0: 32(pad of 5) -> 128
    gemm_kernel<32, 128><<<dim3(BS_ / 64, 1), 256, 0, stream>>>(X0, Wh[br][0], YA);
    stats_kernel<128><<<256, 128, 0, stream>>>(YA, partials);
    finalize_kernel<128><<<1, 128, 0, stream>>>(partials, Gf[br][0], Bef[br][0], normp);
    normrelu_kernel<128><<<(BS_ * 128 / 8) / 256, 256, 0, stream>>>(YA, normp);
    // L1: 128 -> 128
    gemm_kernel<128, 128><<<dim3(BS_ / 64, 1), 256, 0, stream>>>(YA, Wh[br][1], YB);
    stats_kernel<128><<<256, 128, 0, stream>>>(YB, partials);
    finalize_kernel<128><<<1, 128, 0, stream>>>(partials, Gf[br][1], Bef[br][1], normp);
    normrelu_kernel<128><<<(BS_ * 128 / 8) / 256, 256, 0, stream>>>(YB, normp);
    // L2: 128 -> 256, then BN+ReLU fused into channel-max
    gemm_kernel<128, 256><<<dim3(BS_ / 64, 2), 256, 0, stream>>>(YB, Wh[br][2], YC);
    stats_kernel<256><<<256, 256, 0, stream>>>(YC, partials);
    finalize_kernel<256><<<1, 256, 0, stream>>>(partials, Gf[br][2], Bef[br][2], normp);
    maxred_kernel<<<BS_ / 256, 256, 0, stream>>>(YC, normp, br == 0 ? cmax_a : cmax_b);
  }

  // ---- 6) softmax over K + weighted gather fusion ----
  fusion_kernel<<<(B_ * N_) / 8, 256, 0, stream>>>(cmax_a, cmax_b, nnidx, pts2, f2, out);
}
